// TernaryWeight_64854006169958
// MI455X (gfx1250) — matrix operand order repaired and checked by reading, not compiled, not run on hardware
//
#include <hip/hip_runtime.h>
#include <hip/hip_bf16.h>
#include <stdint.h>

#define IN_F   4096
#define OUT_F  11008
#define TOKENS 4096

#define BM 128
#define BN 128
#define BK 64
#define LDK 72   // padded K stride in halves (144 B: 16B-aligned rows, conflict-free)

typedef __attribute__((ext_vector_type(16))) _Float16 v16h;
typedef __attribute__((ext_vector_type(8)))  _Float16 v8h;
typedef __attribute__((ext_vector_type(8)))  float    v8f;
typedef __attribute__((ext_vector_type(4)))  float    v4f;

// ---------------- Pass 1: partial sums of |w| (deterministic fixed-stride) ---------
__global__ __launch_bounds__(256) void absmean_partial(const float* __restrict__ w,
                                                       float* __restrict__ partials,
                                                       long long n4) {
    __shared__ float red[256];
    float s = 0.f;
    long long stride = (long long)gridDim.x * blockDim.x;
    for (long long i = (long long)blockIdx.x * blockDim.x + threadIdx.x; i < n4; i += stride) {
        v4f v = *(const v4f*)(w + i * 4);
        s += fabsf(v.x) + fabsf(v.y) + fabsf(v.z) + fabsf(v.w);
    }
    red[threadIdx.x] = s;
    __syncthreads();
    for (int o = 128; o > 0; o >>= 1) {
        if ((int)threadIdx.x < o) red[threadIdx.x] += red[threadIdx.x + o];
        __syncthreads();
    }
    if (threadIdx.x == 0) partials[blockIdx.x] = red[0];
}

// ---------------- Pass 2: finalize alpha = sum / count (single block, fixed order) --
__global__ __launch_bounds__(256) void absmean_final(const float* __restrict__ partials,
                                                     float* __restrict__ alpha,
                                                     float inv_count) {
    __shared__ float red[256];
    int t = threadIdx.x;
    float s = partials[t] + partials[t + 256] + partials[t + 512] + partials[t + 768];
    red[t] = s;
    __syncthreads();
    for (int o = 128; o > 0; o >>= 1) {
        if (t < o) red[t] += red[t + o];
        __syncthreads();
    }
    if (t == 0) *alpha = red[0] * inv_count;
}

// ---------------- Convert x (fp32) -> f16 ------------------------------------------
__global__ __launch_bounds__(256) void cvt_x_f16(const float* __restrict__ x,
                                                 _Float16* __restrict__ xh) {
    long long i4 = (long long)blockIdx.x * blockDim.x + threadIdx.x;
    v4f v = *(const v4f*)(x + i4 * 4);
    _Float16* p = xh + i4 * 4;
    p[0] = (_Float16)v.x; p[1] = (_Float16)v.y; p[2] = (_Float16)v.z; p[3] = (_Float16)v.w;
}

// ---------------- Ternarize weight, fold scale, emit f16 ---------------------------
__global__ __launch_bounds__(256) void quant_w_f16(const float* __restrict__ w,
                                                   const float* __restrict__ scale,
                                                   const float* __restrict__ alphaPtr,
                                                   _Float16* __restrict__ wh) {
    const float alpha = *alphaPtr;
    long long i4 = (long long)blockIdx.x * blockDim.x + threadIdx.x;
    long long base = i4 * 4;
    const float sc = scale[base / IN_F];   // 4 consecutive elems share a row (IN_F % 4 == 0)
    v4f v = *(const v4f*)(w + base);
    _Float16* p = wh + base;
    #pragma unroll
    for (int j = 0; j < 4; ++j) {
        float e = (j == 0) ? v.x : (j == 1) ? v.y : (j == 2) ? v.z : v.w;
        float q = (e > alpha) ? 1.f : ((e < -alpha) ? -1.f : 0.f);
        p[j] = (_Float16)(q * sc);
    }
}

// ---------------- WMMA GEMM: out = xh @ wh^T + bias --------------------------------
// Block: 128x128 tile, 256 threads = 8 waves (2x4 wave grid), each wave 64x32.
// BK=64 -> 16 v_wmma_f32_16x16x32_f16 per barrier interval.
// Global->LDS staging uses gfx1250 async DMA (GLOBAL_LOAD_ASYNC_TO_LDS_B128,
// tracked by ASYNCcnt), double-buffered.
__global__ __launch_bounds__(256, 1) void gemm_wmma_f16(const _Float16* __restrict__ xh,
                                                        const _Float16* __restrict__ wh,
                                                        const float* __restrict__ bias,
                                                        float* __restrict__ out) {
    __shared__ __attribute__((aligned(16))) _Float16 Ash[2][BM * LDK];
    __shared__ __attribute__((aligned(16))) _Float16 Bsh[2][BN * LDK];
    const uint32_t BUFA = sizeof(_Float16) * BM * LDK;   // 18432 B per A buffer
    const uint32_t BUFB = sizeof(_Float16) * BN * LDK;   // 18432 B per B buffer

    const int tid  = threadIdx.x;
    const int bn   = blockIdx.x;        // 86 N-tiles
    const int bm   = blockIdx.y;        // 32 M-tiles
    const int wid  = tid >> 5;
    const int lane = tid & 31;
    const int hsel = lane >> 4;         // lane-half select (0/1)
    const int l16  = lane & 15;
    const int waveM = (wid >> 2) * 64;  // 0 / 64
    const int waveN = (wid & 3) * 32;   // 0 / 32 / 64 / 96

    // Staging map: tile = 128 rows x 64 halves = 1024 chunks of 8 halves (16 B).
    // Thread handles chunks tid + j*256, j = 0..3.
    uint64_t gA[4], gB[4];
    uint32_t lA[4], lB[4];
    #pragma unroll
    for (int j = 0; j < 4; ++j) {
        int c   = tid + j * 256;
        int row = c >> 3;
        int col = (c & 7) * 8;
        gA[j] = (uint64_t)(uintptr_t)(xh + (size_t)(bm * BM + row) * IN_F + col);
        gB[j] = (uint64_t)(uintptr_t)(wh + (size_t)(bn * BN + row) * IN_F + col);
        lA[j] = (uint32_t)(uintptr_t)&Ash[0][row * LDK + col];
        lB[j] = (uint32_t)(uintptr_t)&Bsh[0][row * LDK + col];
    }

    // Async-issue one K-tile (A+B) into LDS buffer `buf` from byte offset kk*2.
    auto issue_tile = [&](int buf, int kk) {
        const uint64_t go = (uint64_t)(uint32_t)(kk * 2);
        #pragma unroll
        for (int j = 0; j < 4; ++j) {
            uint32_t la = lA[j] + (uint32_t)buf * BUFA;
            uint64_t ga = gA[j] + go;
            asm volatile("global_load_async_to_lds_b128 %0, %1, off"
                         :: "v"(la), "v"(ga) : "memory");
            uint32_t lb = lB[j] + (uint32_t)buf * BUFB;
            uint64_t gb = gB[j] + go;
            asm volatile("global_load_async_to_lds_b128 %0, %1, off"
                         :: "v"(lb), "v"(gb) : "memory");
        }
    };

    v8f acc[4][2];
    #pragma unroll
    for (int mi = 0; mi < 4; ++mi)
        #pragma unroll
        for (int ni = 0; ni < 2; ++ni)
            acc[mi][ni] = (v8f)0.f;

    // Prologue: fill buffer 0.
    issue_tile(0, 0);
    asm volatile("s_wait_asynccnt 0" ::: "memory");
    __syncthreads();

    const int NT = IN_F / BK;           // 64 K-steps
    for (int t = 0; t < NT; ++t) {
        const int cur  = t & 1;
        const bool more = (t + 1) < NT;
        if (more) issue_tile(cur ^ 1, (t + 1) * BK);   // DMA next tile while computing

        #pragma unroll
        for (int ks = 0; ks < BK; ks += 32) {
            // Fragment loads per ISA 16-bit A (16x32) / B (32x16) lane layouts.
            v16h a[4], b[2];
            union U { v16h v; v8h h[2]; };
            #pragma unroll
            for (int mi = 0; mi < 4; ++mi) {
                const _Float16* p = &Ash[cur][(waveM + mi * 16 + l16) * LDK + ks + hsel * 8];
                U u;
                u.h[0] = *(const v8h*)p;          // K = ks + hsel*8 .. +7
                u.h[1] = *(const v8h*)(p + 16);   // K = ks + 16 + hsel*8 .. +7
                a[mi] = u.v;
            }
            #pragma unroll
            for (int ni = 0; ni < 2; ++ni) {
                const _Float16* p = &Bsh[cur][(waveN + ni * 16 + l16) * LDK + ks + hsel * 8];
                U u;
                u.h[0] = *(const v8h*)p;          // K = ks + hsel*8 .. +7
                u.h[1] = *(const v8h*)(p + 16);   // K = ks + 16 + hsel*8 .. +7
                b[ni] = u.v;
            }
            #pragma unroll
            for (int mi = 0; mi < 4; ++mi)
                #pragma unroll
                for (int ni = 0; ni < 2; ++ni)
                    acc[mi][ni] = __builtin_amdgcn_wmma_f32_16x16x32_f16(
                        false, a[mi], false, b[ni], (short)0, acc[mi][ni], false, false);
        }

        if (more) {
            asm volatile("s_wait_asynccnt 0" ::: "memory");  // own DMA writes landed
            __syncthreads();                                  // everyone's landed
        }
    }

    // --- epilogue: C/D layout -> row M = r + hsel*8, col N = l16 ---
    #pragma unroll
    for (int mi = 0; mi < 4; ++mi) {
        #pragma unroll
        for (int ni = 0; ni < 2; ++ni) {
            const int gN = bn * BN + waveN + ni * 16 + l16;
            const float bv = bias[gN];
            #pragma unroll
            for (int r = 0; r < 8; ++r) {
                const int gM = bm * BM + waveM + mi * 16 + hsel * 8 + r;
                out[(size_t)gM * OUT_F + gN] = acc[mi][ni][r] + bv;
            }
        }
    }
}

extern "C" void kernel_launch(void* const* d_in, const int* in_sizes, int n_in,
                              void* d_out, int out_size, void* d_ws, size_t ws_size,
                              hipStream_t stream) {
    const float* x     = (const float*)d_in[0];
    const float* w     = (const float*)d_in[1];
    const float* scale = (const float*)d_in[2];
    const float* bias  = (const float*)d_in[3];
    float* out = (float*)d_out;

    // Workspace layout (needs ~124 MB):
    //   [0, 4KB)          1024 partial sums
    //   [4KB, 8KB)        alpha
    //   [8KB, +32MB)      x as f16
    //   [.., +90MB)       ternarized weight as f16
    char* ws = (char*)d_ws;
    float*     partials = (float*)ws;
    float*     alpha    = (float*)(ws + 4096);
    _Float16*  xh       = (_Float16*)(ws + 8192);
    _Float16*  wh       = (_Float16*)(ws + 8192 + (size_t)TOKENS * IN_F * sizeof(_Float16));

    const long long nW = (long long)OUT_F * IN_F;           // 45,088,768
    absmean_partial<<<1024, 256, 0, stream>>>(w, partials, nW / 4);
    absmean_final<<<1, 256, 0, stream>>>(partials, alpha, (float)(1.0 / (double)nW));

    cvt_x_f16<<<(int)(((long long)TOKENS * IN_F) / 1024), 256, 0, stream>>>(x, xh);
    quant_w_f16<<<(int)(nW / 1024), 256, 0, stream>>>(w, scale, alpha, wh);

    dim3 grid(OUT_F / BN, TOKENS / BM);   // (86, 32)
    gemm_wmma_f16<<<grid, 256, 0, stream>>>(xh, wh, bias, out);
}
